// YatConvTranspose_4346506903788
// MI455X (gfx1250) — matrix operand order repaired and checked by reading, not compiled, not run on hardware
//
#include <hip/hip_runtime.h>
#include <hip/hip_bf16.h>
#include <math.h>

// CDNA5 / gfx1250: wave32, WMMA bf16 16x16x32 with f32 accumulate.
typedef __attribute__((ext_vector_type(16))) __bf16 v16bf;
typedef __attribute__((ext_vector_type(8)))  float  v8f;

#define YAT_EPS 1e-5f

// f32 pair -> packed bf16 dword (RNE), via HW packed convert
__device__ __forceinline__ unsigned pk2(float a, float b) {
  float2 f; f.x = a; f.y = b;
  __hip_bfloat162 h = __float22bfloat162_rn(f);
  unsigned u;
  __builtin_memcpy(&u, &h, 4);
  return u;
}

// ---------------------------------------------------------------------------
// Prep 1: rearranged transposed-conv weights, bf16.
// Bt[cls][co][K] with K = t*256 + ci, t = th*2+tw, kh = py+2*th, kw = px+2*tw.
// ---------------------------------------------------------------------------
__global__ void prep_weights(const float* __restrict__ kern,
                             unsigned short* __restrict__ Bt) {
  int idx = blockIdx.x * 256 + threadIdx.x;   // [0, 4*128*1024)
  int K   = idx & 1023;
  int co  = (idx >> 10) & 127;
  int cls = idx >> 17;
  int t = K >> 8, ci = K & 255;
  int py = cls >> 1, px = cls & 1;
  int kh = py + 2 * (t >> 1);
  int kw = px + 2 * (t & 1);
  float v = kern[(((kh * 4 + kw) * 256 + ci) * 128) + co];
  unsigned u = __float_as_uint(v);
  u = (u + 0x7fffu + ((u >> 16) & 1u)) >> 16;
  Bt[idx] = (unsigned short)u;
}

// ---------------------------------------------------------------------------
// Prep 2: per-filter squared kernel norm k_sq[co] (exact f32).
// ---------------------------------------------------------------------------
__global__ void prep_ksq(const float* __restrict__ kern, float* __restrict__ ksq) {
  int co = threadIdx.x;   // 128 threads, coalesced over co
  float s = 0.f;
  for (int i = 0; i < 4096; ++i) { float v = kern[i * 128 + co]; s += v * v; }
  ksq[co] = s;
}

// ---------------------------------------------------------------------------
// Prep 3: per-input-pixel channel sum of x^2 (exact f32): smap[b*4096+ih*64+iw]
// ---------------------------------------------------------------------------
__global__ void prep_smap(const float* __restrict__ x, float* __restrict__ smap) {
  __shared__ float red[256];
  int tid = threadIdx.x;
  int iw = tid >> 2, q = tid & 3;
  int bh = blockIdx.x;                                  // b*64 + ih
  const float4* p = (const float4*)(x + (size_t)(bh * 64 + iw) * 256 + q * 64);
  float s = 0.f;
#pragma unroll
  for (int i = 0; i < 16; ++i) {
    float4 f = p[i];
    s += f.x * f.x + f.y * f.y + f.z * f.z + f.w * f.w;
  }
  red[tid] = s;
  __syncthreads();
  if (q == 0) smap[bh * 64 + iw] = red[tid] + red[tid + 1] + red[tid + 2] + red[tid + 3];
}

// ---------------------------------------------------------------------------
// Main kernel: per parity class, GEMM  A[256x1024] x B[1024x128] per workgroup.
// 8 wave32, each wave: 32 M-rows (2 A frags) x 128 N -> 16 WMMA / K-step,
// each B fragment reused for 2 WMMAs. Double-buffered LDS, one barrier/step:
// global loads for step n+1 are in flight across step n's WMMA chain.
// grid = 4 classes * 256 WGs, block = 256.
// ---------------------------------------------------------------------------
__global__ __launch_bounds__(256) void yat_main(
    const float* __restrict__ x, const unsigned short* __restrict__ Bt,
    const float* __restrict__ smap, const float* __restrict__ ksq,
    const float* __restrict__ bias, const float* __restrict__ alphap,
    float* __restrict__ out) {
  // LDS tiles: 80B row stride (16B-aligned, reduces b128 bank conflicts)
  __shared__ __align__(16) unsigned char ldsA[2 * 256 * 80]; // A: 256 rows x 32 bf16
  __shared__ __align__(16) unsigned char ldsB[2 * 128 * 80]; // Bt: 128 cols x 32 bf16
  __shared__ float ldsRow[256];                              // per-row patch_sq

  const int tid = threadIdx.x;
  const int cls = blockIdx.x >> 8;       // parity class
  const int w   = blockIdx.x & 255;      // tile id within class
  const int py = cls >> 1, px = cls & 1;
  const int b   = w >> 4;                // batch
  const int oy0 = (w & 15) * 4;          // 4 oy rows x 64 ox = 256 M-rows

  const int soy = oy0 + (tid >> 6);      // this thread's staging pixel
  const int sox = tid & 63;

  // --- patch_sq per output row (exact f32 from smap, OOB taps contribute 0) ---
  {
    float s = 0.f;
#pragma unroll
    for (int t = 0; t < 4; ++t) {
      int ih = soy + py + (t >> 1) - 1;
      int iw = sox + px + (t & 1) - 1;
      if ((unsigned)ih < 64u && (unsigned)iw < 64u)
        s += smap[(b * 64 + ih) * 64 + iw];
    }
    ldsRow[tid] = s;
  }

  v8f acc0[8] = {};   // rows [wv*32, wv*32+16)
  v8f acc1[8] = {};   // rows [wv*32+16, wv*32+32)

  const int lane  = tid & 31;
  const int wv    = tid >> 5;            // wave id: M-strip = 32 rows
  const int khalf = lane >> 4;
  const int arow0 = wv * 32 + (lane & 15);

  // B staging: thread handles col co = tid>>1, half = tid&1 (16 bf16 = 32B)
  const int bco   = tid >> 1;
  const int bhalf = tid & 1;
  const unsigned short* btBase =
      Bt + (((size_t)cls * 128 + bco) << 10) + bhalf * 16;

  // K steps: s = t*8 + kc, K-slice = [s*32, s*32+32)
  float4 f[8];
  uint4  q0, q1;
  auto issue_loads = [&](int s) {
    const int t   = s >> 3;
    const int ci0 = (s & 7) * 32;
    int ih = soy + py + (t >> 1) - 1;
    int iw = sox + px + (t & 1) - 1;
    bool valid = ((unsigned)ih < 64u) && ((unsigned)iw < 64u);
#pragma unroll
    for (int i = 0; i < 8; ++i) f[i] = float4{0.f, 0.f, 0.f, 0.f};
    if (valid) {
      const float4* xp =
          (const float4*)(x + (size_t)((b * 64 + ih) * 64 + iw) * 256 + ci0);
      f[0] = xp[0]; f[1] = xp[1]; f[2] = xp[2]; f[3] = xp[3];
      f[4] = xp[4]; f[5] = xp[5]; f[6] = xp[6]; f[7] = xp[7];
    }
    const uint4* bsrc = (const uint4*)(btBase + s * 32);
    q0 = bsrc[0]; q1 = bsrc[1];
  };

  issue_loads(0);

#pragma unroll 1
  for (int s = 0; s < 32; ++s) {
    unsigned char* aBuf = ldsA + (s & 1) * (256 * 80);
    unsigned char* bBuf = ldsB + (s & 1) * (128 * 80);

    // ---- pack & store step s into LDS buffer (s&1) ----
    uint4 pa, pb, pc, pd;
    pa.x = pk2(f[0].x, f[0].y); pa.y = pk2(f[0].z, f[0].w);
    pa.z = pk2(f[1].x, f[1].y); pa.w = pk2(f[1].z, f[1].w);
    pb.x = pk2(f[2].x, f[2].y); pb.y = pk2(f[2].z, f[2].w);
    pb.z = pk2(f[3].x, f[3].y); pb.w = pk2(f[3].z, f[3].w);
    pc.x = pk2(f[4].x, f[4].y); pc.y = pk2(f[4].z, f[4].w);
    pc.z = pk2(f[5].x, f[5].y); pc.w = pk2(f[5].z, f[5].w);
    pd.x = pk2(f[6].x, f[6].y); pd.y = pk2(f[6].z, f[6].w);
    pd.z = pk2(f[7].x, f[7].y); pd.w = pk2(f[7].z, f[7].w);
    uint4* adst = (uint4*)(aBuf + tid * 80);
    adst[0] = pa; adst[1] = pb; adst[2] = pc; adst[3] = pd;
    uint4* bdst = (uint4*)(bBuf + bco * 80 + bhalf * 32);
    bdst[0] = q0; bdst[1] = q1;

    // ---- issue global loads for step s+1 (in flight during compute) ----
    if (s + 1 < 32) issue_loads(s + 1);

    __syncthreads();   // stores to buffer (s&1) visible; protects buffer reuse

    // ---- A fragments (ISA 16-bit A 16x32 layout):
    //   lane row = lane&15, K dwords {kh*8+0..7} then {16+kh*8+0..7} ----
    v16bf afrag0, afrag1;
    {
      const uint4* ap0 = (const uint4*)(aBuf + arow0 * 80 + khalf * 16);
      ((uint4*)&afrag0)[0] = ap0[0];
      ((uint4*)&afrag0)[1] = ap0[2];                 // +32 bytes
      const uint4* ap1 = (const uint4*)(aBuf + (arow0 + 16) * 80 + khalf * 16);
      ((uint4*)&afrag1)[0] = ap1[0];
      ((uint4*)&afrag1)[1] = ap1[2];
    }
    // ---- B fragments: lane col = lane&15, K = (lane>>4)*8 + 0..7, then 16 + (lane>>4)*8 + 0..7;
    //      each B fragment feeds 2 WMMAs (M reuse) ----
#pragma unroll
    for (int nt = 0; nt < 8; ++nt) {
      const int co = nt * 16 + (lane & 15);
      const uint4* bp = (const uint4*)(bBuf + co * 80 + khalf * 16);
      v16bf bfrag;
      ((uint4*)&bfrag)[0] = bp[0];
      ((uint4*)&bfrag)[1] = bp[2];
      acc0[nt] = __builtin_amdgcn_wmma_f32_16x16x32_bf16(
          false, afrag0, false, bfrag, (short)0, acc0[nt], false, false);
      acc1[nt] = __builtin_amdgcn_wmma_f32_16x16x32_bf16(
          false, afrag1, false, bfrag, (short)0, acc1[nt], false, false);
    }
  }

  // ---- fused Yat epilogue ----
  const float alpha = alphap[0];
  const float scale = __powf(2.3280139f, alpha);  // (sqrt(128)/log1p(128))^a

  float kq[8], bi[8], rs0[8], rs1[8];
#pragma unroll
  for (int nt = 0; nt < 8; ++nt) {
    int co = nt * 16 + (lane & 15);
    kq[nt] = ksq[co];
    bi[nt] = bias[co];
  }
#pragma unroll
  for (int r = 0; r < 8; ++r) {
    rs0[r] = ldsRow[wv * 32 + khalf * 8 + r];
    rs1[r] = ldsRow[wv * 32 + 16 + khalf * 8 + r];
  }

#pragma unroll
  for (int nt = 0; nt < 8; ++nt) {
    const int co = nt * 16 + (lane & 15);
#pragma unroll
    for (int r = 0; r < 8; ++r) {
      {
        const int m = wv * 32 + khalf * 8 + r;
        float dot = acc0[nt][r];
        float dist = rs0[r] + kq[nt] - 2.f * dot;
        float y = ((dot * dot) / (dist + YAT_EPS) + bi[nt]) * scale;
        int oy = oy0 + (m >> 6), ox = m & 63;
        int oh = 2 * oy + py, ow = 2 * ox + px;
        __builtin_nontemporal_store(
            y, out + (((size_t)(b * 128 + oh) * 128 + ow) * 128) + co);
      }
      {
        const int m = wv * 32 + 16 + khalf * 8 + r;
        float dot = acc1[nt][r];
        float dist = rs1[r] + kq[nt] - 2.f * dot;
        float y = ((dot * dot) / (dist + YAT_EPS) + bi[nt]) * scale;
        int oy = oy0 + (m >> 6), ox = m & 63;
        int oh = 2 * oy + py, ow = 2 * ox + px;
        __builtin_nontemporal_store(
            y, out + (((size_t)(b * 128 + oh) * 128 + ow) * 128) + co);
      }
    }
  }
}

// ---------------------------------------------------------------------------
extern "C" void kernel_launch(void* const* d_in, const int* in_sizes, int n_in,
                              void* d_out, int out_size, void* d_ws, size_t ws_size,
                              hipStream_t stream) {
  const float* x     = (const float*)d_in[0];
  const float* kern  = (const float*)d_in[1];
  const float* bias  = (const float*)d_in[2];
  const float* alpha = (const float*)d_in[3];
  float* out = (float*)d_out;

  // workspace layout
  unsigned short* Bt = (unsigned short*)d_ws;                       // 1 MiB
  char* p = (char*)d_ws + (size_t)4 * 128 * 1024 * 2;
  float* smap = (float*)p;                                          // 256 KiB
  float* ksq  = (float*)(p + (size_t)65536 * 4);                    // 512 B

  prep_weights<<<2048, 256, 0, stream>>>(kern, Bt);
  prep_ksq<<<1, 128, 0, stream>>>(kern, ksq);
  prep_smap<<<1024, 256, 0, stream>>>(x, smap);
  yat_main<<<4 * 256, 256, 0, stream>>>(x, Bt, smap, ksq, bias, alpha, out);
}
